// TriangleMultiplication_71519795413184
// MI455X (gfx1250) — matrix operand order repaired and checked by reading, not compiled, not run on hardware
//
#include <hip/hip_runtime.h>
#include <hip/hip_bf16.h>

// MI455X / gfx1250 TriangleMultiplication, bf16 WMMA pipeline.
// ~86 GFLOP; v_wmma_f32_16x16x32_bf16 everywhere, bf16 intermediates kept in
// the 192MB L2, streamed fp32 in/out marked nontemporal. keinsum stages the
// per-(b,i) B^T block in LDS via global_load_async_to_lds_b128 (ASYNCcnt).

typedef __bf16 bf16_t;
typedef __attribute__((ext_vector_type(16))) __bf16 v16bf;
typedef __attribute__((ext_vector_type(8)))  float  v8f;
typedef __attribute__((ext_vector_type(4)))  float  v4f;

union FragBF {
    v16bf v;
    uint4 q[2];
};

// ---- fragment loaders -----------------------------------------------------
// A-matrix 16x32 bf16 fragment (row-major source).
// lanes 0-15 -> rows 0-15, K [0..7] (VGPR0-3) + [8..15] (VGPR4-7);
// lanes 16-31 -> same rows, K [16..23] + [24..31].
__device__ inline v16bf load_fragA_bf16(const bf16_t* base, int stride,
                                        int lane, int kbase) {
    int r = lane & 15, hi = lane >> 4;
    const uint4* p0 = reinterpret_cast<const uint4*>(base + (long)r * stride + kbase + hi * 16);
    const uint4* p1 = reinterpret_cast<const uint4*>(base + (long)r * stride + kbase + 8 + hi * 16);
    FragBF f;
    f.q[0] = p0[0];
    f.q[1] = p1[0];
    return f.v;
}

// Same fragment from fp32 source, nontemporal (pair_repr is streamed once;
// keep it out of L2 so bf16 intermediates stay resident).
__device__ inline v16bf load_fragA_f32_nt(const float* base, int stride,
                                          int lane, int kbase) {
    int r = lane & 15, hi = lane >> 4;
    const v4f* p0 = reinterpret_cast<const v4f*>(base + (long)r * stride + kbase + hi * 16);
    const v4f* p1 = reinterpret_cast<const v4f*>(base + (long)r * stride + kbase + 8 + hi * 16);
    v4f a0 = __builtin_nontemporal_load(p0);
    v4f a1 = __builtin_nontemporal_load(p0 + 1);
    v4f b0 = __builtin_nontemporal_load(p1);
    v4f b1 = __builtin_nontemporal_load(p1 + 1);
    v16bf v;
#pragma unroll
    for (int i = 0; i < 4; ++i) {
        v[i]      = (__bf16)a0[i];
        v[4 + i]  = (__bf16)a1[i];
        v[8 + i]  = (__bf16)b0[i];
        v[12 + i] = (__bf16)b1[i];
    }
    return v;
}

// B-matrix 32x16 bf16 fragment from TRANSPOSED source bt[n][k] (row-major):
// lanes 0-15 hold K=0..15, lanes 16-31 hold K=16..31 -> 32B contiguous/lane.
__device__ inline v16bf load_fragB_bt(const bf16_t* bt, int stride,
                                      int lane, int n0, int kbase) {
    int n  = n0 + (lane & 15);
    int kh = (lane >> 4) * 16;
    const uint4* p = reinterpret_cast<const uint4*>(bt + (long)n * stride + kbase + kh);
    FragBF f;
    f.q[0] = p[0];
    f.q[1] = p[1];
    return f.v;
}

__device__ inline v8f wmma_bf16(v16bf a, v16bf b, v8f c) {
    return __builtin_amdgcn_wmma_f32_16x16x32_bf16(false, a, false, b,
                                                   (short)0, c, false, false);
}

// ---- kernel 0: W -> W^T, fp32 -> bf16 (4 weights) -------------------------
__global__ __launch_bounds__(256) void kprep(const float* __restrict__ Wa,
                                             const float* __restrict__ Wb,
                                             const float* __restrict__ Wg,
                                             const float* __restrict__ Wo,
                                             bf16_t* __restrict__ wt) {
    int idx  = blockIdx.x * 256 + threadIdx.x;   // 4*256*256 total
    int widx = idx >> 16;
    int rem  = idx & 65535;
    int k    = rem >> 8;
    int n    = rem & 255;
    const float* W = (widx == 0) ? Wa : (widx == 1) ? Wb : (widx == 2) ? Wg : Wo;
    wt[(widx << 16) + n * 256 + k] = (bf16_t)W[k * 256 + n];
}

// ---- kernel 1: fused triple projection ------------------------------------
// 256 rows per block, two 16-row strips per wave: each B fragment feeds two
// WMMAs. A fragments (fp32->bf16) converted once, reused across 3 weights.
__global__ __launch_bounds__(256) void kproj(const float* __restrict__ x,
                                             const bf16_t* __restrict__ wt,
                                             const float* __restrict__ ba,
                                             const float* __restrict__ bb,
                                             const float* __restrict__ bg,
                                             bf16_t* __restrict__ aws,
                                             bf16_t* __restrict__ btws,
                                             bf16_t* __restrict__ gws) {
    int lane = threadIdx.x & 31;
    int wave = threadIdx.x >> 5;
    long row0 = (long)blockIdx.x * 256 + wave * 16;   // strip 0
    long row1 = row0 + 128;                           // strip 1

    v16bf fa0[8], fa1[8];
#pragma unroll
    for (int kk = 0; kk < 8; ++kk) {
        fa0[kk] = load_fragA_f32_nt(x + row0 * 256, 256, lane, kk * 32);
        fa1[kk] = load_fragA_f32_nt(x + row1 * 256, 256, lane, kk * 32);
    }

#pragma unroll
    for (int widx = 0; widx < 3; ++widx) {
        const bf16_t* wtp  = wt + widx * 65536;
        const float*  bias = (widx == 0) ? ba : (widx == 1) ? bb : bg;
        for (int ct = 0; ct < 16; ++ct) {
            v8f acc0 = {}, acc1 = {};
#pragma unroll
            for (int kk = 0; kk < 8; ++kk) {
                v16bf fb = load_fragB_bt(wtp, 256, lane, ct * 16, kk * 32);
                acc0 = wmma_bf16(fa0[kk], fb, acc0);
                acc1 = wmma_bf16(fa1[kk], fb, acc1);
            }
            int   col = ct * 16 + (lane & 15);
            float bv  = bias[col];
#pragma unroll
            for (int v = 0; v < 8; ++v) {
                int  m  = ((lane >> 4) << 3) + v;
                long R0 = row0 + m, R1 = row1 + m;
                float v0 = acc0[v] + bv, v1 = acc1[v] + bv;
                if (widx == 0) {                      // a: row-major bf16
                    aws[R0 * 256 + col] = (bf16_t)v0;
                    aws[R1 * 256 + col] = (bf16_t)v1;
                } else if (widx == 1) {               // b: transposed per (b,i)
                    btws[((R0 >> 8) << 16) + (long)col * 256 + (R0 & 255)] = (bf16_t)v0;
                    btws[((R1 >> 8) << 16) + (long)col * 256 + (R1 & 255)] = (bf16_t)v1;
                } else {                              // g: sigmoid, row-major
                    gws[R0 * 256 + col] = (bf16_t)__builtin_amdgcn_rcpf(1.0f + __expf(-v0));
                    gws[R1 * 256 + col] = (bf16_t)__builtin_amdgcn_rcpf(1.0f + __expf(-v1));
                }
            }
        }
    }
}

// ---- kernel 2: per-(b,i) einsum GEMM + in-place gating --------------------
// B^T block (128KB) staged in LDS via async-to-LDS copies; B fragments then
// come from ds_load_b128 with a 528B padded row stride (conflict-free).
// Each wave runs two strips so each LDS B fragment feeds two WMMAs.
__global__ __launch_bounds__(256) void keinsum(const bf16_t* __restrict__ aws,
                                               const bf16_t* __restrict__ btws,
                                               bf16_t* __restrict__ gws) {
    __shared__ __align__(16) unsigned char sbt[256 * 528];   // 135168 B

    int lane = threadIdx.x & 31;
    int wave = threadIdx.x >> 5;
    long bi  = blockIdx.x;                       // 0..511
    const bf16_t* A = aws + (bi << 16);
    bf16_t*       G = gws + (bi << 16);

    {   // async stage: 8192 x 16B chunks, 32 per thread (ASYNCcnt <= 32)
        const bf16_t* BTg = btws + (bi << 16);
        for (int c = threadIdx.x; c < 8192; c += 256) {
            int row    = c >> 5;                 // n-row of B^T
            int within = c & 31;                 // 16B chunk within 512B row
            unsigned lds_off = (unsigned)(uintptr_t)(sbt + row * 528 + within * 16);
            unsigned goff    = (unsigned)c * 16u;
            asm volatile("global_load_async_to_lds_b128 %0, %1, %2"
                         :: "v"(lds_off), "v"(goff), "s"(BTg) : "memory");
        }
        asm volatile("s_wait_asynccnt 0x0" ::: "memory");
    }
    __syncthreads();

    int r0 = wave * 16;                          // strips r0 and r0+128
    v16bf fa0[8], fa1[8];
#pragma unroll
    for (int kk = 0; kk < 8; ++kk) {
        fa0[kk] = load_fragA_bf16(A + (long)r0 * 256, 256, lane, kk * 32);
        fa1[kk] = load_fragA_bf16(A + (long)(r0 + 128) * 256, 256, lane, kk * 32);
    }

    for (int ct = 0; ct < 16; ++ct) {
        v8f acc0 = {}, acc1 = {};
        int n = ct * 16 + (lane & 15);
#pragma unroll
        for (int kk = 0; kk < 8; ++kk) {
            int k2 = kk * 32 + (lane >> 4) * 16;
            const uint4* p = reinterpret_cast<const uint4*>(sbt + n * 528 + k2 * 2);
            FragBF f;
            f.q[0] = p[0];
            f.q[1] = p[1];
            acc0 = wmma_bf16(fa0[kk], f.v, acc0);
            acc1 = wmma_bf16(fa1[kk], f.v, acc1);
        }
        int col = ct * 16 + (lane & 15);
#pragma unroll
        for (int v = 0; v < 8; ++v) {
            int  m0 = r0 + ((lane >> 4) << 3) + v;
            long i0 = (long)m0 * 256 + col;
            long i1 = (long)(m0 + 128) * 256 + col;
            G[i0] = (bf16_t)((float)G[i0] * acc0[v]);   // gate in place
            G[i1] = (bf16_t)((float)G[i1] * acc1[v]);
        }
    }
}

// ---- kernel 3: output projection ------------------------------------------
// out = gated @ Wo + bo  (bf16 in, fp32 NT stores: output never re-read).
__global__ __launch_bounds__(256) void kout(const bf16_t* __restrict__ gated,
                                            const bf16_t* __restrict__ wt,
                                            const float* __restrict__ bo,
                                            float* __restrict__ out) {
    int lane = threadIdx.x & 31;
    int wave = threadIdx.x >> 5;
    long row0 = (long)blockIdx.x * 256 + wave * 16;
    long row1 = row0 + 128;
    const bf16_t* wto = wt + 3 * 65536;

    v16bf fa0[8], fa1[8];
#pragma unroll
    for (int kk = 0; kk < 8; ++kk) {
        fa0[kk] = load_fragA_bf16(gated + row0 * 256, 256, lane, kk * 32);
        fa1[kk] = load_fragA_bf16(gated + row1 * 256, 256, lane, kk * 32);
    }

    for (int ct = 0; ct < 16; ++ct) {
        v8f acc0 = {}, acc1 = {};
#pragma unroll
        for (int kk = 0; kk < 8; ++kk) {
            v16bf fb = load_fragB_bt(wto, 256, lane, ct * 16, kk * 32);
            acc0 = wmma_bf16(fa0[kk], fb, acc0);
            acc1 = wmma_bf16(fa1[kk], fb, acc1);
        }
        int   col = ct * 16 + (lane & 15);
        float bv  = bo[col];
#pragma unroll
        for (int v = 0; v < 8; ++v) {
            int m = ((lane >> 4) << 3) + v;
            __builtin_nontemporal_store(acc0[v] + bv, &out[(row0 + m) * 256 + col]);
            __builtin_nontemporal_store(acc1[v] + bv, &out[(row1 + m) * 256 + col]);
        }
    }
}

extern "C" void kernel_launch(void* const* d_in, const int* in_sizes, int n_in,
                              void* d_out, int out_size, void* d_ws, size_t ws_size,
                              hipStream_t stream) {
    const float* x  = (const float*)d_in[0];   // pair_repr [2,256,256,256]
    const float* Wa = (const float*)d_in[1];
    const float* ba = (const float*)d_in[2];
    const float* Wb = (const float*)d_in[3];
    const float* bb = (const float*)d_in[4];
    const float* Wg = (const float*)d_in[5];
    const float* bg = (const float*)d_in[6];
    const float* Wo = (const float*)d_in[7];
    const float* bo = (const float*)d_in[8];

    // workspace: [0,512KB) W^T bf16 | a (64MB) | b^T (64MB) | g->gated (64MB)
    bf16_t* wt   = (bf16_t*)d_ws;
    bf16_t* aws  = (bf16_t*)((char*)d_ws + 524288);
    bf16_t* btws = aws + 33554432;
    bf16_t* gws  = btws + 33554432;

    kprep  <<<1024, 256, 0, stream>>>(Wa, Wb, Wg, Wo, wt);
    kproj  <<< 512, 256, 0, stream>>>(x, wt, ba, bb, bg, aws, btws, gws);
    keinsum<<< 512, 256, 0, stream>>>(aws, btws, gws);
    kout   <<< 512, 256, 0, stream>>>(gws, wt, bo, (float*)d_out);
}
